// CFPT_68925635167016
// MI455X (gfx1250) — matrix operand order repaired and checked by reading, not compiled, not run on hardware
//
#include <hip/hip_runtime.h>

typedef __bf16 bf16_t;
typedef bf16_t v16bf __attribute__((ext_vector_type(16)));
typedef bf16_t bf16x4 __attribute__((ext_vector_type(4)));
typedef float  v8f   __attribute__((ext_vector_type(8)));
typedef float  f32x4 __attribute__((ext_vector_type(4)));
typedef unsigned int u32x4 __attribute__((ext_vector_type(4)));

#define PB 264      /* bf16 pitch for token-major 64x256 mats (8-elem pad: bank-conflict free) */
#define PS 66       /* f32 pitch for 64x64 score rows */
#define PV 72       /* bf16 pitch for transposed V (256 x 64) and for P (64 x 64) */
#define OFF_X 0          /* X bf16, later reused as attention-output O bf16   (33792 B) */
#define OFF_S 33792      /* Y bf16, later reused as f32 scores (2 heads)      (33792 B) */
#define OFF_Q 67584      /* Q bf16                                            (33792 B) */
#define OFF_K 101376     /* K bf16                                            (33792 B) */
#define OFF_V 135168     /* V^T bf16, 256 rows x 72 pitch                     (36864 B) */
#define OFF_P 172032     /* P bf16 (softmax probs), 2 heads x 64 x 72 pitch   (18432 B) */
#define SMEM_BYTES 190464

// native hardware f32 -> bf16 convert (RNE on gfx1250)
__device__ __forceinline__ bf16_t f2bf(float x) { return (bf16_t)x; }

// A-matrix fragment (16x32 bf16): lane holds row m, K = {kh..kh+7, kh+16..kh+23}
// caller passes p = &M[row][k0 + kh] with kh = 8*(lane>>4)
__device__ __forceinline__ v16bf ld_fragA(const bf16_t* p) {
  union { u32x4 u[2]; v16bf v; } f;
  f.u[0] = *(const u32x4*)(p);
  f.u[1] = *(const u32x4*)(p + 16);
  return f.v;
}

// B-matrix fragment (32x16 bf16): lane holds col n, K = {kh..kh+7, kh+16..kh+23}
// (same K order as A); p = &Bcol[n][k0 + khB], khB = 8*(lane>>4)
__device__ __forceinline__ v16bf ld_fragB(const bf16_t* p) {
  union { u32x4 u[2]; v16bf v; } f;
  f.u[0] = *(const u32x4*)(p);
  f.u[1] = *(const u32x4*)(p + 16);
  return f.v;
}

#define WMMA_BF16(acc, a, b) \
  __builtin_amdgcn_wmma_f32_16x16x32_bf16(false, (a), false, (b), (short)0, (acc), false, false)

__global__ void cvt_bf16_kernel(const float* __restrict__ src, bf16_t* __restrict__ dst, int n4) {
  int i = blockIdx.x * 256 + threadIdx.x;
  if (i < n4) {
    f32x4 v = *(const f32x4*)(src + 4 * i);
    bf16x4 o;
    o.x = f2bf(v.x); o.y = f2bf(v.y); o.z = f2bf(v.z); o.w = f2bf(v.w);
    *(bf16x4*)(dst + 4 * i) = o;
  }
}

// One workgroup (8 wave32) = one 8x8 window. Full cross-attention:
//   out = relu(x + Wo^T(softmax(scale * (Wq x)(Wk y)^T) (Wv y)))
__global__ __launch_bounds__(256, 1)
void swin_attn_kernel(const float* __restrict__ xp, const float* __restrict__ yp,
                      float* __restrict__ op, const bf16_t* __restrict__ Wm,
                      int H, int W, int nwin) {
  extern __shared__ char smem[];
  bf16_t* Xs = (bf16_t*)(smem + OFF_X);   // X, later O (pre-Wo attention output)
  bf16_t* Ys = (bf16_t*)(smem + OFF_S);   // Y
  float*  Ss = (float*)(smem + OFF_S);    // scores (reuses Y space)
  bf16_t* Qs = (bf16_t*)(smem + OFF_Q);
  bf16_t* Ks = (bf16_t*)(smem + OFF_K);
  bf16_t* Vt = (bf16_t*)(smem + OFF_V);   // V transposed: [d][token]
  bf16_t* Ps = (bf16_t*)(smem + OFF_P);   // softmax probs bf16: [head][q][k]

  const int tid  = threadIdx.x;
  const int lane = tid & 31;
  const int wv   = tid >> 5;
  const int lr   = lane & 15;             // fragment row (A) / col (B) / col (D)
  const int khA  = (lane >> 4) << 3;      // A-fragment K base offset (0 or 8)
  const int khB  = (lane >> 4) << 3;      // B-fragment K base offset (0 or 8)

  const int g  = blockIdx.x;
  const int b  = g / nwin;
  const int wi = g - b * nwin;
  const int wc = W >> 3;
  const int h0 = (wi / wc) << 3;
  const int w0 = (wi - (wi / wc) * wc) << 3;

  // ---- stage X and Y windows (f32 NCHW -> bf16 token-major LDS), float4 chunks ----
  for (int e = tid; e < 4096; e += 256) {
    int c  = e >> 4;                      // channel
    int t0 = (e & 15) << 2;               // tokens t0..t0+3 (same window row: 4 | 8)
    size_t gi = (((size_t)b * 256 + c) * H + (h0 + (t0 >> 3))) * W + (w0 + (t0 & 7));
    f32x4 xv = *(const f32x4*)(xp + gi);
    f32x4 yv = *(const f32x4*)(yp + gi);
#pragma unroll
    for (int j = 0; j < 4; ++j) {
      Xs[(t0 + j) * PB + c] = f2bf(xv[j]);
      Ys[(t0 + j) * PB + c] = f2bf(yv[j]);
    }
  }
  __syncthreads();

  // ---- Q = X Wq^T, K = Y Wk^T, V = Y Wv^T  (192 tiles of 16x16, K=256) ----
  for (int tile = wv; tile < 192; tile += 8) {
    const int which = tile >> 6;          // 0:Q 1:K 2:V (wave-uniform)
    const int tt = tile & 63;
    const int mt = tt >> 4, nt = tt & 15;
    const bf16_t* A  = (which == 0) ? Xs : Ys;
    const bf16_t* Wp = Wm + which * 65536;  // Wq/Wk/Wv, row-major (d,c) bf16
    v8f acc = {};
#pragma unroll
    for (int kt = 0; kt < 8; ++kt) {
      v16bf a  = ld_fragA(A  + (mt * 16 + lr) * PB  + kt * 32 + khA);
      v16bf bb = ld_fragB(Wp + (nt * 16 + lr) * 256 + kt * 32 + khB);
      acc = WMMA_BF16(acc, a, bb);
    }
    const int mb = mt * 16 + khA;         // D rows: lanes 0-15 -> M 0..7, 16-31 -> M 8..15
    if (which == 2) {
      const int d = nt * 16 + lr;
#pragma unroll
      for (int v = 0; v < 8; ++v) Vt[d * PV + mb + v] = f2bf(acc[v]);
    } else {
      bf16_t* D = (which == 0) ? Qs : Ks;
#pragma unroll
      for (int v = 0; v < 8; ++v) D[(mb + v) * PB + nt * 16 + lr] = f2bf(acc[v]);
    }
  }
  __syncthreads();

  // ---- scores: S[h] = scale * Q_h K_h^T  (2 heads x 4x4 tiles, K=128) ----
  const float scale = 0.088388347648318447f;  // 128^-0.5
  for (int tile = wv; tile < 32; tile += 8) {
    const int hd = tile >> 4;
    const int mt = (tile >> 2) & 3;
    const int nt = tile & 3;
    v8f acc = {};
#pragma unroll
    for (int kt = 0; kt < 4; ++kt) {
      v16bf a  = ld_fragA(Qs + (mt * 16 + lr) * PB + hd * 128 + kt * 32 + khA);
      v16bf bb = ld_fragB(Ks + (nt * 16 + lr) * PB + hd * 128 + kt * 32 + khB);
      acc = WMMA_BF16(acc, a, bb);
    }
    float* S = Ss + hd * 64 * PS;
    const int mb = mt * 16 + khA;
#pragma unroll
    for (int v = 0; v < 8; ++v) S[(mb + v) * PS + nt * 16 + lr] = acc[v] * scale;
  }
  __syncthreads();

  // ---- softmax: one thread per (head,row); emit bf16 probs into Ps ----
  if (tid < 128) {
    const int hd = tid >> 6, r = tid & 63;
    float* row = Ss + hd * 64 * PS + r * PS;
    bf16_t* prow = Ps + hd * 64 * PV + r * PV;
    float m = row[0];
    for (int j = 1; j < 64; ++j) m = fmaxf(m, row[j]);
    float s = 0.f;
    for (int j = 0; j < 64; ++j) { float e = __expf(row[j] - m); row[j] = e; s += e; }
    float inv = 1.f / s;
    for (int j = 0; j < 64; ++j) prow[j] = f2bf(row[j] * inv);
  }
  __syncthreads();

  // ---- O = P V  (per head 4x8 tiles, K=64); store into Xs (X is dead) ----
  for (int tile = wv; tile < 64; tile += 8) {
    const int hd = tile >> 5;
    const int mt = (tile >> 3) & 3;
    const int nt = tile & 7;
    v8f acc = {};
#pragma unroll
    for (int kt = 0; kt < 2; ++kt) {
      v16bf a  = ld_fragA(Ps + hd * 64 * PV + (mt * 16 + lr) * PV + kt * 32 + khA);
      v16bf bb = ld_fragB(Vt + (hd * 128 + nt * 16 + lr) * PV + kt * 32 + khB);
      acc = WMMA_BF16(acc, a, bb);
    }
    const int mb = mt * 16 + khA;
#pragma unroll
    for (int v = 0; v < 8; ++v)
      Xs[(mb + v) * PB + hd * 128 + nt * 16 + lr] = f2bf(acc[v]);
  }
  __syncthreads();

  // ---- final: out = relu(x + O Wo^T), vectorized NCHW writeback ----
  const bf16_t* Wo = Wm + 3 * 65536;
  for (int tile = wv; tile < 64; tile += 8) {
    const int mt = tile >> 4, nt = tile & 15;
    v8f acc = {};
#pragma unroll
    for (int kt = 0; kt < 8; ++kt) {
      v16bf a  = ld_fragA(Xs + (mt * 16 + lr) * PB  + kt * 32 + khA);
      v16bf bb = ld_fragB(Wo + (nt * 16 + lr) * 256 + kt * 32 + khB);
      acc = WMMA_BF16(acc, a, bb);
    }
    const int mb = mt * 16 + khA;         // multiple of 8 -> 8 tokens in one window row
    const int c  = nt * 16 + lr;
    size_t gi = (((size_t)b * 256 + c) * H + (h0 + (mb >> 3))) * W + w0;
    f32x4 x0 = *(const f32x4*)(xp + gi);
    f32x4 x1 = *(const f32x4*)(xp + gi + 4);
    f32x4 r0, r1;
#pragma unroll
    for (int v = 0; v < 4; ++v) {
      float a0 = x0[v] + acc[v];     r0[v] = a0 > 0.f ? a0 : 0.f;
      float a1 = x1[v] + acc[v + 4]; r1[v] = a1 > 0.f ? a1 : 0.f;
    }
    *(f32x4*)(op + gi)     = r0;
    *(f32x4*)(op + gi + 4) = r1;
  }
}

extern "C" void kernel_launch(void* const* d_in, const int* in_sizes, int n_in,
                              void* d_out, int out_size, void* d_ws, size_t ws_size,
                              hipStream_t stream) {
  (void)in_sizes; (void)n_in; (void)out_size; (void)ws_size;
  const float* w1 = (const float*)d_in[4];
  const float* w2 = (const float*)d_in[5];
  bf16_t* wbf = (bf16_t*)d_ws;
  const int NW = 4 * 4 * 256 * 256;  // elements per weight tensor (multiple of 4)

  cvt_bf16_kernel<<<NW / 4 / 256, 256, 0, stream>>>(w1, wbf, NW / 4);
  cvt_bf16_kernel<<<NW / 4 / 256, 256, 0, stream>>>(w2, wbf + NW, NW / 4);

  float* out = (float*)d_out;
  const int Hs[4] = {128, 64, 32, 16};
  size_t off = 0;
  for (int i = 0; i < 4; ++i) {
    const int H = Hs[i];
    const size_t half = (size_t)2 * 256 * H * H;    // l=2 batches per half
    const float* fin = (const float*)d_in[i];
    float* lo = out + off;                          // f1' destination
    float* hi = out + off + half;                   // f2' destination
    const int nwin = (H / 8) * (H / 8);
    const dim3 grid(2 * nwin);
    const bf16_t* Wm1 = wbf + (size_t)(i * 4) * 65536;
    const bf16_t* Wm2 = wbf + NW + (size_t)(i * 4) * 65536;

    // f1' = relu(f1 + attn(f1, f2; w1[i]))
    swin_attn_kernel<<<grid, 256, SMEM_BYTES, stream>>>(fin, fin + half, lo, Wm1, H, H, nwin);
    // f2' = relu(f2 + attn(f2, f1'; w2[i]))  -- reads attn1's output (stream-ordered)
    swin_attn_kernel<<<grid, 256, SMEM_BYTES, stream>>>(fin + half, lo, hi, Wm2, H, H, nwin);

    off += 2 * half;
  }
}
